// Comm_OUT_13537736917304
// MI455X (gfx1250) — compile-verified, matrix operand order checked by reading, not run on hardware
//
#include <hip/hip_runtime.h>
#include <hip/hip_bf16.h>
#include <stdint.h>

// ---------------------------------------------------------------------------
// CDNA5 (gfx1250) wave32 WMMA implementation of the Comm_OUT pipeline.
// GEMMs run on v_wmma_f32_16x16x32_f16 (f16 in, f32 accumulate).
// Weights are kept in their native [N,K] row-major layout: that is exactly the
// per-lane B-fragment layout (lane n holds K-pairs of W row n), so every
// fragment register-pair is a single aligned b128 load (compiler coalesces).
// Each wave register-blocks a 32x64 output tile: 2 A frags x 4 B frags ->
// 8 WMMAs per 32-wide K chunk (0.75 fragment loads per WMMA).
// ---------------------------------------------------------------------------

typedef __attribute__((ext_vector_type(16))) _Float16 v16h;
typedef __attribute__((ext_vector_type(8)))  float    v8f;

union FragU { v16h v; uint32_t u[8]; };

// Load a 16x32 f16 fragment (A layout, ISA 7.12.2) from a row-major [rows, ld]
// matrix. Works for both A tiles and W ([N,K] row-major == B^T) tiles.
// Per-lane K pattern: lanes 0-15 -> K {0..7, 16..23}, lanes 16-31 -> +8.
__device__ __forceinline__ v16h load_frag16(const _Float16* __restrict__ base,
                                            int ld, int row0, int k0, int lane) {
  const int half = (lane >> 4) & 1;
  const int r    = row0 + (lane & 15);
  const uint32_t* q = (const uint32_t*)(base + (size_t)r * ld + k0 + half * 8);
  FragU f;
  f.u[0] = q[0];  f.u[1] = q[1];  f.u[2] = q[2];  f.u[3] = q[3];
  f.u[4] = q[8];  f.u[5] = q[9];  f.u[6] = q[10]; f.u[7] = q[11];
  return f.v;
}

// ---------------------------------------------------------------------------
// Templated WMMA GEMM:  D[M,N] = A[M,K] @ W[N,K]^T  (+ epilogue)
//   MODE 0: out32[m*N+n] = acc + bias[n]
//   MODE 1: out16[m*N+n] = prelu(bn(acc + bias[n]))      (fused BN + PReLU)
//   MODE 2: out32[(b*L+l)*C+n] = acc + bias[n]  where m = l*B + b  (head remap)
// Each wave computes a (16*MB) x (16*NB) tile; K looped in chunks of 32.
// ---------------------------------------------------------------------------
template <int MODE, int MB, int NB>
__global__ __launch_bounds__(256) void gemm_wmma_f16(
    const _Float16* __restrict__ A, const _Float16* __restrict__ W,
    int M, int N, int K,
    const float* __restrict__ bias,
    const float* __restrict__ gam, const float* __restrict__ bet,
    const float* __restrict__ mean, const float* __restrict__ var,
    const float* __restrict__ alpha,
    float* __restrict__ out32, _Float16* __restrict__ out16,
    int Ldim, int Bdim, int Cdim) {
  const int lane  = threadIdx.x & 31;
  const int wave  = blockIdx.x * (blockDim.x >> 5) + (threadIdx.x >> 5);
  const int ntn   = N / (16 * NB);
  const int total = (M / (16 * MB)) * ntn;
  if (wave >= total) return;
  const int mt = wave / ntn;
  const int nt = wave - mt * ntn;
  const int m0 = mt * (16 * MB);
  const int n0 = nt * (16 * NB);

  v8f acc[MB][NB] = {};

  for (int k0 = 0; k0 < K; k0 += 32) {
    v16h a[MB];
#pragma unroll
    for (int i = 0; i < MB; ++i) a[i] = load_frag16(A, K, m0 + i * 16, k0, lane);
#pragma unroll
    for (int j = 0; j < NB; ++j) {
      v16h b = load_frag16(W, K, n0 + j * 16, k0, lane);
#pragma unroll
      for (int i = 0; i < MB; ++i) {
        // D = A(16x32) x B(32x16) + C, f32 accumulate
        acc[i][j] = __builtin_amdgcn_wmma_f32_16x16x32_f16(
            /*neg_a=*/false, a[i], /*neg_b=*/false, b,
            /*c_mod=*/(short)0, acc[i][j], /*reuse_a=*/false, /*reuse_b=*/false);
      }
    }
  }

  // C/D layout: VGPR r -> row (tile_m0) + (lane>>4)*8 + r ; column n0 + (lane&15)
  const int ncol = lane & 15;
  float al = 0.0f;
  if constexpr (MODE == 1) al = alpha[0];

#pragma unroll
  for (int j = 0; j < NB; ++j) {
    const int n = n0 + j * 16 + ncol;
    const float bi = bias[n];
    float sc = 1.0f, sh = 0.0f;
    if constexpr (MODE == 1) {
      sc = gam[n] * rsqrtf(var[n] + 1e-5f);
      sh = bet[n] - mean[n] * sc;
    }
#pragma unroll
    for (int i = 0; i < MB; ++i) {
      const int mrow0 = m0 + i * 16 + ((lane >> 4) << 3);
#pragma unroll
      for (int r = 0; r < 8; ++r) {
        const int m = mrow0 + r;
        float v = acc[i][j][r] + bi;
        if constexpr (MODE == 1) {
          v = v * sc + sh;                 // eval-mode BatchNorm1d
          v = (v >= 0.0f) ? v : al * v;    // PReLU (scalar alpha)
          out16[(size_t)m * N + n] = (_Float16)v;
        } else if constexpr (MODE == 2) {
          const int bb = m % Bdim;         // m = l*B + b
          const int ll = m / Bdim;
          out32[((size_t)bb * Ldim + ll) * Cdim + n] = v;
        } else {
          out32[(size_t)m * N + n] = v;
        }
      }
    }
  }
}

// ---------------------------------------------------------------------------
// GRU gate fusion: consumes gi (precomputed, loop-invariant) and gh (this
// step), updates h (f32 state + f16 GEMM-input copy), and stores the
// BN2+PReLU'd hidden state directly as the f16 A-matrix for the conv GEMM.
// ---------------------------------------------------------------------------
__global__ __launch_bounds__(256) void gru_gate_kernel(
    const float* __restrict__ gi, const float* __restrict__ gh,
    float* __restrict__ h32, _Float16* __restrict__ h16,
    _Float16* __restrict__ hsy,
    const float* __restrict__ g2, const float* __restrict__ be2,
    const float* __restrict__ m2, const float* __restrict__ v2,
    const float* __restrict__ a2,
    int Bn, int Hn, int l) {
  const int i = blockIdx.x * blockDim.x + threadIdx.x;
  const int n = Bn * Hn;
  if (i >= n) return;
  const int b = i / Hn;
  const int c = i - b * Hn;
  const int G = 3 * Hn;
  const size_t base = (size_t)b * G + c;

  const float ir = gi[base], iz = gi[base + Hn], inn = gi[base + 2 * Hn];
  const float hr = gh[base], hz = gh[base + Hn], hnn = gh[base + 2 * Hn];

  const float r  = 1.0f / (1.0f + __expf(-(ir + hr)));
  const float z  = 1.0f / (1.0f + __expf(-(iz + hz)));
  const float nn = tanhf(inn + r * hnn);
  const float h  = (1.0f - z) * nn + z * h32[i];

  h32[i] = h;
  h16[i] = (_Float16)h;

  // fused BN2 + PReLU into the stored sequence (eval-mode, elementwise)
  const float sc = g2[c] * rsqrtf(v2[c] + 1e-5f);
  float y = (h - m2[c]) * sc + be2[c];
  y = (y >= 0.0f) ? y : a2[0] * y;
  hsy[(size_t)l * n + i] = (_Float16)y;
}

__global__ void cvt_f32_f16(const float* __restrict__ s, _Float16* __restrict__ d, int n) {
  const int i = blockIdx.x * blockDim.x + threadIdx.x;
  if (i < n) d[i] = (_Float16)s[i];
}

__global__ void init_h_kernel(float* __restrict__ h32, _Float16* __restrict__ h16, int n) {
  const int i = blockIdx.x * blockDim.x + threadIdx.x;
  if (i < n) { h32[i] = 0.0f; h16[i] = (_Float16)0.0f; }
}

// ---------------------------------------------------------------------------

extern "C" void kernel_launch(void* const* d_in, const int* in_sizes, int n_in,
                              void* d_out, int out_size, void* d_ws, size_t ws_size,
                              hipStream_t stream) {
  (void)in_sizes; (void)n_in; (void)out_size; (void)ws_size;
  const int E = 64, S = 128, F = 640, H = 256, C = 32, L = 20;
  const int B = E * S;        // 8192
  const int G = 3 * H;        // 768
  const int BL = B * L;       // 163840

  // inputs in setup_inputs() dict order
  const float* h_w_action = (const float*)d_in[0];
  const float* W_lin = (const float*)d_in[1];
  const float* b_lin = (const float*)d_in[2];
  const float* g1 = (const float*)d_in[3];
  const float* beta1 = (const float*)d_in[4];
  const float* m1 = (const float*)d_in[5];
  const float* v1 = (const float*)d_in[6];
  const float* a1 = (const float*)d_in[7];
  const float* W_ih = (const float*)d_in[8];
  const float* W_hh = (const float*)d_in[9];
  const float* b_ih = (const float*)d_in[10];
  const float* b_hh = (const float*)d_in[11];
  const float* g2 = (const float*)d_in[12];
  const float* beta2 = (const float*)d_in[13];
  const float* m2 = (const float*)d_in[14];
  const float* v2 = (const float*)d_in[15];
  const float* a2 = (const float*)d_in[16];
  const float* Wc = (const float*)d_in[17];
  const float* bc = (const float*)d_in[18];
  const float* g3 = (const float*)d_in[19];
  const float* beta3 = (const float*)d_in[20];
  const float* m3 = (const float*)d_in[21];
  const float* v3 = (const float*)d_in[22];
  const float* a3 = (const float*)d_in[23];
  const float* W_mu = (const float*)d_in[24];
  const float* b_mu = (const float*)d_in[25];

  // carve workspace (aligned 256B)
  char* p = (char*)d_ws;
  auto carve = [&](size_t bytes) -> void* {
    void* r = (void*)p;
    p += (bytes + 255) & ~(size_t)255;
    return r;
  };
  _Float16* Xh     = (_Float16*)carve((size_t)B * F * 2);   // input, f16
  _Float16* Wlin_h = (_Float16*)carve((size_t)H * F * 2);
  _Float16* Wih_h  = (_Float16*)carve((size_t)G * H * 2);
  _Float16* Whh_h  = (_Float16*)carve((size_t)G * H * 2);
  _Float16* Wc_h   = (_Float16*)carve((size_t)H * H * 2);
  _Float16* Wmu_h  = (_Float16*)carve((size_t)C * H * 2);
  _Float16* x_h    = (_Float16*)carve((size_t)B * H * 2);   // post BN1+PReLU
  float*    gi     = (float*)carve((size_t)B * G * 4);      // loop-invariant gates
  float*    gh     = (float*)carve((size_t)B * G * 4);      // per-step gates
  float*    h32    = (float*)carve((size_t)B * H * 4);      // GRU state f32
  _Float16* h16    = (_Float16*)carve((size_t)B * H * 2);   // GRU state f16 (GEMM A)
  _Float16* hsy    = (_Float16*)carve((size_t)BL * H * 2);  // prelu(bn2(hs)) [L,B,H]
  _Float16* y2h    = (_Float16*)carve((size_t)BL * H * 2);  // prelu(bn3(conv)) [L*B,H]

  auto cvt = [&](const float* s, _Float16* d, int n) {
    cvt_f32_f16<<<(n + 255) / 256, 256, 0, stream>>>(s, d, n);
  };
  cvt(h_w_action, Xh, B * F);
  cvt(W_lin, Wlin_h, H * F);
  cvt(W_ih, Wih_h, G * H);
  cvt(W_hh, Whh_h, G * H);
  cvt(Wc, Wc_h, H * H);
  cvt(W_mu, Wmu_h, C * H);

  init_h_kernel<<<(B * H + 255) / 256, 256, 0, stream>>>(h32, h16, B * H);

  // Stage 1: x = prelu(bn1(X @ W_lin^T + b_lin)) -> f16   [8192,640]x[640,256]
  {
    const int waves = (B / 32) * (H / 64);      // 1024
    gemm_wmma_f16<1, 2, 4><<<waves / 8, 256, 0, stream>>>(
        Xh, Wlin_h, B, H, F, b_lin, g1, beta1, m1, v1, a1,
        nullptr, x_h, 0, 0, 0);
  }
  // Stage 2: gi = x @ W_ih^T + b_ih  (loop-invariant)     [8192,256]x[256,768]
  {
    const int waves = (B / 32) * (G / 64);      // 3072
    gemm_wmma_f16<0, 2, 4><<<waves / 8, 256, 0, stream>>>(
        x_h, Wih_h, B, G, H, b_ih, nullptr, nullptr, nullptr, nullptr, nullptr,
        gi, nullptr, 0, 0, 0);
  }
  // Stage 3: GRU recurrence, L sequential (GEMM + fused gate) pairs.
  // All working state (W_hh 0.4MB, h 4MB, gi/gh 50MB) is L2-resident.
  for (int l = 0; l < L; ++l) {
    const int waves = (B / 32) * (G / 64);      // 3072
    gemm_wmma_f16<0, 2, 4><<<waves / 8, 256, 0, stream>>>(
        h16, Whh_h, B, G, H, b_hh, nullptr, nullptr, nullptr, nullptr, nullptr,
        gh, nullptr, 0, 0, 0);
    gru_gate_kernel<<<(B * H + 255) / 256, 256, 0, stream>>>(
        gi, gh, h32, h16, hsy, g2, beta2, m2, v2, a2, B, H, l);
  }
  // Stage 4: y2 = prelu(bn3(hsy @ Wc^T + bc)) -> f16      [163840,256]x[256,256]
  {
    const int waves = (BL / 32) * (H / 64);     // 20480
    gemm_wmma_f16<1, 2, 4><<<waves / 8, 256, 0, stream>>>(
        hsy, Wc_h, BL, H, H, bc, g3, beta3, m3, v3, a3,
        nullptr, y2h, 0, 0, 0);
  }
  // Stage 5: pred = y2 @ W_mu^T + b_mu, remapped [L,B,C] row -> [B,L,C] output
  {
    const int waves = (BL / 32) * (C / 32);     // 5120
    gemm_wmma_f16<2, 2, 2><<<waves / 8, 256, 0, stream>>>(
        y2h, Wmu_h, BL, C, H, b_mu, nullptr, nullptr, nullptr, nullptr, nullptr,
        (float*)d_out, nullptr, L, B, C);
  }
}
